// CrossAttention_27565100106019
// MI455X (gfx1250) — compile-verified, matrix operand order checked by reading, not run on hardware
//
#include <hip/hip_runtime.h>
#include <hip/hip_bf16.h>
#include <math.h>

typedef __attribute__((ext_vector_type(16))) __bf16 v16bf;
typedef __attribute__((ext_vector_type(8)))  __bf16 v8bf;
typedef __attribute__((ext_vector_type(8)))  float  v8f;
typedef __attribute__((ext_vector_type(4)))  float  v4f;

#define DIMX 512
#define NTOT 65536
#define SCALE_QK 0.125f
#define LN_EPS 1e-5f

__device__ __forceinline__ v8f zero8() {
  v8f z;
#pragma unroll
  for (int i = 0; i < 8; i++) z[i] = 0.f;
  return z;
}

__device__ __forceinline__ v8f wmma_bf16(v16bf a, v16bf b, v8f c) {
  return __builtin_amdgcn_wmma_f32_16x16x32_bf16(false, a, false, b, (short)0, c, false, false);
}

// A fragment (row0 = tile row base) or B fragment (row0 = tile col base, matrix = Y with B=Y^T)
// from a row-major bf16 matrix with leading dim ld. Lane hf = lane>>4, ix = lane&15.
// Lane holds K offsets {k0+hf*8 .. +7} and {k0+16+hf*8 .. +7}.
__device__ __forceinline__ v16bf frag_bf16(const __bf16* base, int row0, size_t ld, int k0,
                                           int hf, int ix) {
  const __bf16* p = base + (size_t)(row0 + ix) * ld + k0 + hf * 8;
  v8bf a = *(const v8bf*)p;
  v8bf b = *(const v8bf*)(p + 16);
  v16bf r;
#pragma unroll
  for (int j = 0; j < 8; j++) { r[j] = a[j]; r[j + 8] = b[j]; }
  return r;
}

__device__ __forceinline__ v16bf frag_f32(const float* base, int row0, size_t ld, int k0,
                                          int hf, int ix) {
  const float* p = base + (size_t)(row0 + ix) * ld + k0 + hf * 8;
  v4f a0 = *(const v4f*)p,        a1 = *(const v4f*)(p + 4);
  v4f b0 = *(const v4f*)(p + 16), b1 = *(const v4f*)(p + 20);
  v16bf r;
#pragma unroll
  for (int j = 0; j < 4; j++) {
    r[j]      = (__bf16)a0[j];
    r[j + 4]  = (__bf16)a1[j];
    r[j + 8]  = (__bf16)b0[j];
    r[j + 12] = (__bf16)b1[j];
  }
  return r;
}

// ---------------- kernel 0: fp32 -> bf16 weight conversion ----------------
__global__ void k_prep(const float* __restrict__ W, const float* __restrict__ Wq,
                       const float* __restrict__ Wgp, __bf16* __restrict__ Wbf,
                       __bf16* __restrict__ Wqbf, __bf16* __restrict__ Wgpbf) {
  int i = blockIdx.x * blockDim.x + threadIdx.x;
  if (i < DIMX * DIMX) {
    Wbf[i]   = (__bf16)W[i];
    Wqbf[i]  = (__bf16)Wq[i];
    Wgpbf[i] = (__bf16)Wgp[i];
  }
}

// ---------------- kernel 0b: g fp32 -> bf16 (row-major copy) ----------------
__global__ void k_cvt_g(const float* __restrict__ g, __bf16* __restrict__ gb) {
  size_t i = ((size_t)blockIdx.x * 256 + threadIdx.x) * 8;
  v4f a = *(const v4f*)(g + i), b = *(const v4f*)(g + i + 4);
  v8bf r;
#pragma unroll
  for (int j = 0; j < 4; j++) { r[j] = (__bf16)a[j]; r[j + 4] = (__bf16)b[j]; }
  *(v8bf*)(gb + i) = r;
}

// ---------------- kernel 0c: g_p fp32 [N][D] -> gpT bf16 [D][N] (LDS-tiled) ----------------
__global__ void k_tr_gp(const float* __restrict__ gp, __bf16* __restrict__ gpT) {
  __shared__ float tile[64][65];
  int t = threadIdx.x;
  int n0 = blockIdx.x * 64, d0 = blockIdx.y * 64;
  {
    int nl = t & 63, dq = t >> 6;  // each thread: row nl, 16 consecutive d
    const float* src = gp + (size_t)(n0 + nl) * DIMX + d0 + dq * 16;
    v4f x0 = *(const v4f*)src, x1 = *(const v4f*)(src + 4);
    v4f x2 = *(const v4f*)(src + 8), x3 = *(const v4f*)(src + 12);
#pragma unroll
    for (int j = 0; j < 4; j++) {
      tile[nl][dq * 16 + j]      = x0[j];
      tile[nl][dq * 16 + 4 + j]  = x1[j];
      tile[nl][dq * 16 + 8 + j]  = x2[j];
      tile[nl][dq * 16 + 12 + j] = x3[j];
    }
  }
  __syncthreads();
  {
    int dl = t & 63, nq = t >> 6;  // each thread: out row d0+dl, 16 consecutive n
    v8bf r0, r1;
#pragma unroll
    for (int j = 0; j < 8; j++) {
      r0[j] = (__bf16)tile[nq * 16 + j][dl];
      r1[j] = (__bf16)tile[nq * 16 + 8 + j][dl];
    }
    __bf16* dst = gpT + (size_t)(d0 + dl) * NTOT + n0 + nq * 16;
    *(v8bf*)dst = r0;
    *(v8bf*)(dst + 8) = r1;
  }
}

// ---------------- kernel 1: pool row-max over N (W tile staged in LDS) ----------------
__global__ void __launch_bounds__(128, 1)
k_pool_max(const __bf16* __restrict__ gb, const __bf16* __restrict__ Wbf,
           float* __restrict__ stats) {
  extern __shared__ char smem[];
  int lane = threadIdx.x & 31, wid = threadIdx.x >> 5;
  int hf = lane >> 4, ix = lane & 15;
  int m0 = (blockIdx.x * 4 + wid) * 16;
  __bf16* Wl = (__bf16*)(smem + wid * 16384);   // 16 rows x 512 bf16 per wave
  for (int e = lane; e < 1024; e += 32) {
    int r = e >> 6, cc = (e & 63) * 8;
    *(v8bf*)(Wl + r * DIMX + cc) = *(const v8bf*)(Wbf + (size_t)(m0 + r) * DIMX + cc);
  }
  __syncthreads();

  float pmax[8];
#pragma unroll
  for (int v = 0; v < 8; v++) pmax[v] = -3.0e38f;
  for (int nb = 0; nb < NTOT; nb += 32) {
    __syncthreads();   // code-motion fence: keep LDS A-frag loads inside the loop
    {
      size_t pn = (size_t)((nb + 32 + lane < NTOT) ? (nb + 32 + lane) : (NTOT - 1));
      __builtin_prefetch(gb + pn * DIMX, 0, 1);
    }
    v8f s0 = zero8(), s1 = zero8();
    for (int kt = 0; kt < 16; kt++) {
      int k0 = kt * 32;
      v16bf a  = frag_bf16(Wl, 0, DIMX, k0, hf, ix);
      v16bf b0 = frag_bf16(gb, nb, DIMX, k0, hf, ix);
      v16bf b1 = frag_bf16(gb, nb + 16, DIMX, k0, hf, ix);
      s0 = wmma_bf16(a, b0, s0);
      s1 = wmma_bf16(a, b1, s1);
    }
#pragma unroll
    for (int v = 0; v < 8; v++)
      pmax[v] = fmaxf(pmax[v], fmaxf(s0[v], s1[v]));
  }
#pragma unroll
  for (int v = 0; v < 8; v++) {
    float mx = pmax[v];
    mx = fmaxf(mx, __shfl_xor(mx, 1, 16));
    mx = fmaxf(mx, __shfl_xor(mx, 2, 16));
    mx = fmaxf(mx, __shfl_xor(mx, 4, 16));
    mx = fmaxf(mx, __shfl_xor(mx, 8, 16));
    if (ix == 0) stats[m0 + v + hf * 8] = mx;
  }
}

// ---------------- kernel 2: k_p = softmax(W g^T) @ g_p (W tile shared in LDS) ----------------
__global__ void __launch_bounds__(128, 1)
k_kp(const __bf16* __restrict__ gb, const __bf16* __restrict__ gpT,
     const __bf16* __restrict__ Wbf, const float* __restrict__ stats,
     float* __restrict__ kp_raw) {
  extern __shared__ char smem[];
  int lane = threadIdx.x & 31, wid = threadIdx.x >> 5;
  int hf = lane >> 4, ix = lane & 15;
  int m0 = blockIdx.x * 16;
  int d0 = wid * 128;
  __bf16* Wl = (__bf16*)smem;                        // shared 16x512 W tile
  __bf16* Pl = (__bf16*)(smem + 16384 + wid * 1024); // per-wave 16x32 P tile
  for (int e = threadIdx.x; e < 1024; e += 128) {
    int r = e >> 6, cc = (e & 63) * 8;
    *(v8bf*)(Wl + r * DIMX + cc) = *(const v8bf*)(Wbf + (size_t)(m0 + r) * DIMX + cc);
  }
  __syncthreads();

  float rm[8], rowsum[8];
#pragma unroll
  for (int v = 0; v < 8; v++) { rm[v] = stats[m0 + v + hf * 8]; rowsum[v] = 0.f; }
  v8f acc[8];
#pragma unroll
  for (int t = 0; t < 8; t++) acc[t] = zero8();

  for (int nb = 0; nb < NTOT; nb += 32) {
    {
      size_t pn = (size_t)((nb + 32 + lane < NTOT) ? (nb + 32 + lane) : (NTOT - 1));
      __builtin_prefetch(gb + pn * DIMX, 0, 1);
    }
    v8f s0 = zero8(), s1 = zero8();
    for (int kt = 0; kt < 16; kt++) {
      int k0 = kt * 32;
      v16bf a  = frag_bf16(Wl, 0, DIMX, k0, hf, ix);
      v16bf b0 = frag_bf16(gb, nb, DIMX, k0, hf, ix);
      v16bf b1 = frag_bf16(gb, nb + 16, DIMX, k0, hf, ix);
      s0 = wmma_bf16(a, b0, s0);
      s1 = wmma_bf16(a, b1, s1);
    }
    // P = exp(S - rowmax): D-layout -> LDS so it can be re-read in A-layout;
    // accumulate lane-local row sums (merged once at the end).
#pragma unroll
    for (int v = 0; v < 8; v++) {
      int row = v + hf * 8;
      float p0 = __expf(s0[v] - rm[v]);
      float p1 = __expf(s1[v] - rm[v]);
      Pl[row * 32 + ix]      = (__bf16)p0;
      Pl[row * 32 + 16 + ix] = (__bf16)p1;
      rowsum[v] += p0 + p1;
    }
    __syncthreads();
    v16bf af;
    {
      const __bf16* p = Pl + ix * 32 + hf * 8;
#pragma unroll
      for (int j = 0; j < 8; j++) { af[j] = p[j]; af[j + 8] = p[16 + j]; }
    }
#pragma unroll
    for (int t = 0; t < 8; t++) {
      v16bf bf_ = frag_bf16(gpT, d0 + t * 16, (size_t)NTOT, nb, hf, ix);
      acc[t] = wmma_bf16(af, bf_, acc[t]);
    }
    __syncthreads();
  }
  float inv[8];
#pragma unroll
  for (int v = 0; v < 8; v++) {
    float sm = rowsum[v];
    sm += __shfl_xor(sm, 1, 16);
    sm += __shfl_xor(sm, 2, 16);
    sm += __shfl_xor(sm, 4, 16);
    sm += __shfl_xor(sm, 8, 16);
    inv[v] = 1.f / sm;
  }
#pragma unroll
  for (int t = 0; t < 8; t++) {
    int dcol = d0 + t * 16;
#pragma unroll
    for (int v = 0; v < 8; v++) {
      int row = m0 + v + hf * 8;
      kp_raw[(size_t)row * DIMX + dcol + ix] = acc[t][v] * inv[v];
    }
  }
}

// ---------------- kernel 3: K = Ws Wk^T + bk + kp Wkp^T + bkp ; V^T ----------------
__global__ void k_buildKV(const float* __restrict__ Wk, const float* __restrict__ bk,
                          const float* __restrict__ Wv, const float* __restrict__ bv,
                          const float* __restrict__ Wkp, const float* __restrict__ bkp,
                          const __bf16* __restrict__ Wbf, const float* __restrict__ kp_raw,
                          __bf16* __restrict__ Kbf, __bf16* __restrict__ Vtbf) {
  int lane = threadIdx.x & 31, wid = threadIdx.x >> 5;
  int hf = lane >> 4, ix = lane & 15;
  int tid = blockIdx.x * 4 + wid;   // 1024 16x16 tiles
  int m0 = (tid & 31) * 16;
  int d0 = (tid >> 5) * 16;
  float biasK = bk[d0 + ix] + bkp[d0 + ix];
  float biasV = bv[d0 + ix];
  v8f accK, accV;
#pragma unroll
  for (int v = 0; v < 8; v++) { accK[v] = biasK; accV[v] = biasV; }
  for (int kt = 0; kt < 16; kt++) {
    int k0 = kt * 32;
    v16bf aW   = frag_bf16(Wbf, m0, DIMX, k0, hf, ix);
    v16bf aKp  = frag_f32(kp_raw, m0, DIMX, k0, hf, ix);
    v16bf bWk  = frag_f32(Wk, d0, DIMX, k0, hf, ix);
    v16bf bWkp = frag_f32(Wkp, d0, DIMX, k0, hf, ix);
    v16bf bWv  = frag_f32(Wv, d0, DIMX, k0, hf, ix);
    accK = wmma_bf16(aW, bWk, accK);
    accK = wmma_bf16(aKp, bWkp, accK);
    accV = wmma_bf16(aW, bWv, accV);
  }
#pragma unroll
  for (int v = 0; v < 8; v++) {
    int row = m0 + v + hf * 8, col = d0 + ix;
    Kbf[(size_t)row * DIMX + col]  = (__bf16)accK[v];
    Vtbf[(size_t)col * DIMX + row] = (__bf16)accV[v];  // store V transposed
  }
}

// ---------------- kernel 4: fused Q proj + scores + softmax + attn@V + residual ----------------
__global__ void __launch_bounds__(128, 1)
k_main(const __bf16* __restrict__ gb, const float* __restrict__ g,
       const float* __restrict__ gp,
       const float* __restrict__ bq, const float* __restrict__ bgp,
       const __bf16* __restrict__ Wqbf, const __bf16* __restrict__ Wgpbf,
       const __bf16* __restrict__ Kbf, const __bf16* __restrict__ Vtbf,
       float* __restrict__ out) {
  extern __shared__ char smem[];
  int lane = threadIdx.x & 31, wid = threadIdx.x >> 5;
  int hf = lane >> 4, ix = lane & 15;
  int n0 = (blockIdx.x * 4 + wid) * 16;
  const int QBYTES = 16 * DIMX * 2;       // Q tile (bf16); later reused as P tile (bf16)
  const int SBYTES = 16 * DIMX * 4;       // phase A: staged g/gp rows (bf16); phase B: scores f32
  const int WBYTES = QBYTES + SBYTES + 128;
  char* base = smem + wid * WBYTES;
  __bf16* Qw = (__bf16*)base;             // phase A/B: Q ; phase C: P
  float*  Sw = (float*)(base + QBYTES);
  float*  st = (float*)(base + QBYTES + SBYTES);
  __bf16* gA  = (__bf16*)(base + QBYTES);                 // phase A only
  __bf16* gpA = (__bf16*)(base + QBYTES + 16 * DIMX * 2); // phase A only

  // Phase 0: stage this wave's 16 g rows (bf16) and 16 g_p rows (cvt fp32->bf16) into LDS.
  for (int e = lane; e < 1024; e += 32) {
    int r = e >> 6, cc = (e & 63) * 8;
    *(v8bf*)(gA + r * DIMX + cc) = *(const v8bf*)(gb + (size_t)(n0 + r) * DIMX + cc);
    const float* ps = gp + (size_t)(n0 + r) * DIMX + cc;
    v4f x0 = *(const v4f*)ps, x1 = *(const v4f*)(ps + 4);
    v8bf rr;
#pragma unroll
    for (int j = 0; j < 4; j++) { rr[j] = (__bf16)x0[j]; rr[j + 4] = (__bf16)x1[j]; }
    *(v8bf*)(gpA + r * DIMX + cc) = rr;
  }
  __syncthreads();

  // Phase A: Q = g Wq^T + gp Wgp^T + bq + bgp  ->  LDS (bf16)
  for (int dt = 0; dt < 32; dt++) {
    int dcol = dt * 16;
    float bia = bq[dcol + ix] + bgp[dcol + ix];
    v8f acc;
#pragma unroll
    for (int v = 0; v < 8; v++) acc[v] = bia;
    for (int kt = 0; kt < 16; kt++) {
      int k0 = kt * 32;
      v16bf ag  = frag_bf16(gA, 0, DIMX, k0, hf, ix);
      v16bf bw  = frag_bf16(Wqbf, dcol, DIMX, k0, hf, ix);
      acc = wmma_bf16(ag, bw, acc);
      v16bf agp = frag_bf16(gpA, 0, DIMX, k0, hf, ix);
      v16bf bw2 = frag_bf16(Wgpbf, dcol, DIMX, k0, hf, ix);
      acc = wmma_bf16(agp, bw2, acc);
    }
#pragma unroll
    for (int v = 0; v < 8; v++)
      Qw[(v + hf * 8) * DIMX + dcol + ix] = (__bf16)acc[v];
  }
  __syncthreads();

  // Phase B: S = Q K^T * SCALE -> LDS; lane-local running max
  float runmax[8];
#pragma unroll
  for (int v = 0; v < 8; v++) runmax[v] = -3.0e38f;
  for (int mt = 0; mt < 32; mt++) {
    int mcol = mt * 16;
    v8f acc = zero8();
    for (int kt = 0; kt < 16; kt++) {
      int k0 = kt * 32;
      v16bf aq  = frag_bf16(Qw, 0, DIMX, k0, hf, ix);
      v16bf bk_ = frag_bf16(Kbf, mcol, DIMX, k0, hf, ix);
      acc = wmma_bf16(aq, bk_, acc);
    }
#pragma unroll
    for (int v = 0; v < 8; v++) {
      float s = acc[v] * SCALE_QK;
      Sw[(v + hf * 8) * DIMX + mcol + ix] = s;
      runmax[v] = fmaxf(runmax[v], s);
    }
  }
  __syncthreads();
#pragma unroll
  for (int v = 0; v < 8; v++) {
    float mx = runmax[v];
    mx = fmaxf(mx, __shfl_xor(mx, 1, 16));
    mx = fmaxf(mx, __shfl_xor(mx, 2, 16));
    mx = fmaxf(mx, __shfl_xor(mx, 4, 16));
    mx = fmaxf(mx, __shfl_xor(mx, 8, 16));
    float sum = 0.f;
    for (int m = ix; m < DIMX; m += 16)
      sum += __expf(Sw[(v + hf * 8) * DIMX + m] - mx);
    sum += __shfl_xor(sum, 1, 16);
    sum += __shfl_xor(sum, 2, 16);
    sum += __shfl_xor(sum, 4, 16);
    sum += __shfl_xor(sum, 8, 16);
    if (ix == 0) {
      st[2 * (v + hf * 8)]     = mx;
      st[2 * (v + hf * 8) + 1] = 1.f / sum;
    }
  }
  __syncthreads();

  // Materialize P = exp(S - max) / sum as bf16 into the dead Q buffer (once).
#pragma unroll
  for (int v = 0; v < 8; v++) {
    int row = v + hf * 8;
    float mx = st[2 * row], rinv = st[2 * row + 1];
#pragma unroll
    for (int j = 0; j < 32; j++) {
      int m = ix + 16 * j;
      Qw[row * DIMX + m] = (__bf16)(__expf(Sw[row * DIMX + m] - mx) * rinv);
    }
  }
  __syncthreads();

  // Phase C: out = P @ V + g, in 4 d-chunks of 128
  for (int c = 0; c < 4; c++) {
    __syncthreads();   // code-motion fence: keep P-frag LDS loads inside the chunk loop
    int dbase = c * 128;
    v8f acc[8];
#pragma unroll
    for (int t = 0; t < 8; t++) acc[t] = zero8();
    for (int kt = 0; kt < 16; kt++) {
      int km0 = kt * 32;
      v16bf af = frag_bf16(Qw, 0, DIMX, km0, hf, ix);
#pragma unroll
      for (int t = 0; t < 8; t++) {
        int dcol = dbase + t * 16;
        v16bf bf_ = frag_bf16(Vtbf, dcol, DIMX, km0, hf, ix);
        acc[t] = wmma_bf16(af, bf_, acc[t]);
      }
    }
#pragma unroll
    for (int t = 0; t < 8; t++) {
      int dcol = dbase + t * 16;
#pragma unroll
      for (int v = 0; v < 8; v++) {
        size_t o = (size_t)(n0 + v + hf * 8) * DIMX + dcol + ix;
        out[o] = acc[t][v] + g[o];
      }
    }
  }
}

// ---------------- kernel 5: layernorm, one wave per row ----------------
__global__ void k_ln(float* __restrict__ out, const float* __restrict__ gamma,
                     const float* __restrict__ beta) {
  int lane = threadIdx.x & 31, wid = threadIdx.x >> 5;
  size_t row = (size_t)blockIdx.x * 8 + wid;
  float* p = out + row * DIMX;
  float x[16], s = 0.f, ss = 0.f;
#pragma unroll
  for (int j = 0; j < 16; j++) {
    x[j] = p[lane + 32 * j];
    s += x[j];
    ss += x[j] * x[j];
  }
#pragma unroll
  for (int m = 1; m < 32; m <<= 1) {
    s += __shfl_xor(s, m, 32);
    ss += __shfl_xor(ss, m, 32);
  }
  float mean = s * (1.f / DIMX);
  float var = ss * (1.f / DIMX) - mean * mean;
  float rstd = rsqrtf(var + LN_EPS);
#pragma unroll
  for (int j = 0; j < 16; j++) {
    int d = lane + 32 * j;
    p[d] = (x[j] - mean) * rstd * gamma[d] + beta[d];
  }
}

extern "C" void kernel_launch(void* const* d_in, const int* in_sizes, int n_in,
                              void* d_out, int out_size, void* d_ws, size_t ws_size,
                              hipStream_t stream) {
  (void)in_sizes; (void)n_in; (void)out_size; (void)ws_size;
  const float* g    = (const float*)d_in[0];
  const float* gp   = (const float*)d_in[1];
  const float* W    = (const float*)d_in[2];
  const float* Wq   = (const float*)d_in[3];
  const float* bq   = (const float*)d_in[4];
  const float* Wk   = (const float*)d_in[5];
  const float* bk   = (const float*)d_in[6];
  const float* Wv   = (const float*)d_in[7];
  const float* bv   = (const float*)d_in[8];
  const float* Wgp  = (const float*)d_in[9];
  const float* bgp  = (const float*)d_in[10];
  const float* Wkp  = (const float*)d_in[11];
  const float* bkp  = (const float*)d_in[12];
  const float* gam  = (const float*)d_in[13];
  const float* bet  = (const float*)d_in[14];
  float* out = (float*)d_out;

  char* ws = (char*)d_ws;
  __bf16* Wbf    = (__bf16*)(ws);                  // 512 KB
  __bf16* Wqbf   = (__bf16*)(ws + 524288);         // 512 KB
  __bf16* Wgpbf  = (__bf16*)(ws + 1048576);        // 512 KB
  __bf16* Kbf    = (__bf16*)(ws + 1572864);        // 512 KB
  __bf16* Vtbf   = (__bf16*)(ws + 2097152);        // 512 KB
  float*  kp_raw = (float*)(ws + 2621440);         // 1 MB
  float*  stats  = (float*)(ws + 3670016);         // 2 KB
  __bf16* gb     = (__bf16*)(ws + 4194304);        // 64 MB: g in bf16
  __bf16* gpT    = (__bf16*)(ws + 71303168);       // 64 MB: g_p^T in bf16

  k_prep<<<1024, 256, 0, stream>>>(W, Wq, Wgp, Wbf, Wqbf, Wgpbf);
  k_cvt_g<<<16384, 256, 0, stream>>>(g, gb);
  dim3 trg(NTOT / 64, DIMX / 64);
  k_tr_gp<<<trg, 256, 0, stream>>>(gp, gpT);
  k_pool_max<<<8, 128, 65536, stream>>>(gb, Wbf, stats);
  k_kp<<<32, 128, 20480, stream>>>(gb, gpT, Wbf, stats, kp_raw);
  k_buildKV<<<256, 128, 0, stream>>>(Wk, bk, Wv, bv, Wkp, bkp, Wbf, kp_raw, Kbf, Vtbf);
  size_t smem = (size_t)4 * (16 * DIMX * 2 + 16 * DIMX * 4 + 128);
  k_main<<<1024, 128, smem, stream>>>(gb, g, gp, bq, bgp, Wqbf, Wgpbf, Kbf, Vtbf, out);
  k_ln<<<8192, 256, 0, stream>>>(out, gam, bet);
}
